// MonotonicNN_50010599194772
// MI455X (gfx1250) — compile-verified, matrix operand order checked by reading, not run on hardware
//
#include <hip/hip_runtime.h>

typedef _Float16 v16h __attribute__((ext_vector_type(16)));
typedef _Float16 v8h  __attribute__((ext_vector_type(8)));
typedef _Float16 v2h  __attribute__((ext_vector_type(2)));
typedef float    v8f  __attribute__((ext_vector_type(8)));

#define NB_STEPS 50
#define S_PTS    51
#define HID      128
#define ST_W     136   // padded halfword stride (272B -> lane bank stride 4, conflict-free)
#define ST_A     136
#define WAVES    8
#define TPB      256
#define TILES_PER_WAVE 8
#define PI_F     3.14159265358979323846f

// out[b] = h[b]  (quadrature atomics then accumulate z on top)
__global__ void __launch_bounds__(256) monotonic_init(const float* __restrict__ h,
                                                      float* __restrict__ out, int n) {
    int i = blockIdx.x * blockDim.x + threadIdx.x;
    if (i < n) out[i] = h[i];
}

// Load a 16x32 f16 A/B fragment row: lane holds K = {klo..klo+7, 16+klo..16+klo+7}
__device__ __forceinline__ v16h ldfrag(const _Float16* rowp, int klo) {
    v8h lo = *(const v8h*)(rowp + klo);
    v8h hi = *(const v8h*)(rowp + 16 + klo);
    return __builtin_shufflevector(lo, hi, 0,1,2,3,4,5,6,7,8,9,10,11,12,13,14,15);
}

__global__ void __launch_bounds__(TPB) monotonic_main(
    const float* __restrict__ x,  const float* __restrict__ h,
    const float* __restrict__ w1, const float* __restrict__ b1,
    const float* __restrict__ w2, const float* __restrict__ b2,
    const float* __restrict__ w3, const float* __restrict__ b3,
    const float* __restrict__ w4, const float* __restrict__ b4,
    float* __restrict__ out, int Bn)
{
    __shared__ __align__(16) _Float16 w2T[HID * ST_W];           // [n][k] f16, transposed
    __shared__ __align__(16) _Float16 w3T[HID * ST_W];           // [n][k] f16, transposed
    __shared__ __align__(16) _Float16 acts[WAVES * 16 * ST_A];   // per-wave 16x128 staging
    __shared__ float w1s[2 * HID], b1s[HID], b2s[HID], b3s[HID], w4s[HID];
    __shared__ float ccws[S_PTS], stepss[S_PTS], b4s[1];

    const int tid = threadIdx.x;

    // ---- cooperative setup (once per block): f16 transposed weights + params ----
    for (int idx = tid; idx < HID * HID; idx += TPB) {
        int k = idx >> 7, n = idx & 127;
        w2T[n * ST_W + k] = (_Float16)w2[idx];
        w3T[n * ST_W + k] = (_Float16)w3[idx];
    }
    if (tid < HID) {
        b1s[tid] = b1[tid]; b2s[tid] = b2[tid]; b3s[tid] = b3[tid]; w4s[tid] = w4[tid];
        w1s[tid] = w1[tid]; w1s[HID + tid] = w1[HID + tid];
    }
    if (tid == 0) b4s[0] = b4[0];
    if (tid < S_PTS) {  // Clenshaw-Curtis weights + abscissae, computed in-kernel
        float t = (float)tid;
        stepss[tid] = cosf(t * PI_F / (float)NB_STEPS);
        float a = 1.0f;
        #pragma unroll
        for (int j = 1; j <= NB_STEPS / 2; ++j) {
            float k = 2.0f * (float)j;
            a += (2.0f / (1.0f - k * k)) * cosf(k * t * PI_F / (float)NB_STEPS);
        }
        a *= 2.0f / (float)NB_STEPS;
        if (tid == 0 || tid == NB_STEPS) a *= 0.5f;
        ccws[tid] = a;
    }
    __syncthreads();   // only barrier; persistent loop below has none

    const int  lane  = tid & 31;
    const int  wave  = tid >> 5;
    const long total   = (long)Bn * S_PTS;
    const long tstride = (long)gridDim.x * WAVES * 16;

    _Float16* act = &acts[wave * 16 * ST_A];
    const int  m   = lane & 15;           // row within tile (A) / col within N-tile (B)
    const int  klo = (lane < 16) ? 0 : 8;
    const int  mo  = (lane >> 4) * 8;

    // ---- persistent grid-stride loop over 16-row tiles (one tile per wave) ----
    for (long tbase = ((long)blockIdx.x * WAVES + wave) * 16; tbase < total; tbase += tstride) {
        const long r  = tbase + m;
        const long rc = (r < total) ? r : (total - 1);
        const int  bidx = (int)(rc / S_PTS);
        const int  sidx = (int)(rc - (long)bidx * S_PTS);
        const float xb = x[bidx];
        const float hb = h[bidx];
        const float Xv = xb * (stepss[sidx] + 1.0f) * 0.5f;

        // ---- layer 1: [X,H] @ w1 + b1, ReLU -> f16 staging (packed b32 stores) ----
        {
            const int khalf = (lane >> 4) * 64;  // lanes 0-15: cols 0-63; 16-31: 64-127
            v2h* actrow = (v2h*)&act[m * ST_A + khalf];   // 4B aligned (khalf even, 272B rows)
            #pragma unroll 4
            for (int kk = 0; kk < 32; ++kk) {
                int k = khalf + 2 * kk;
                float v0 = fmaf(Xv, w1s[k],     fmaf(hb, w1s[HID + k],     b1s[k]));
                float v1 = fmaf(Xv, w1s[k + 1], fmaf(hb, w1s[HID + k + 1], b1s[k + 1]));
                v2h p; p.x = (_Float16)fmaxf(v0, 0.0f); p.y = (_Float16)fmaxf(v1, 0.0f);
                actrow[kk] = p;
            }
        }

        // ---- layer 2: act @ w2 + b2, ReLU (WMMA, B frags double-buffered) ----
        {
            v16h a0 = ldfrag(&act[m * ST_A +  0], klo);
            v16h a1 = ldfrag(&act[m * ST_A + 32], klo);
            v16h a2 = ldfrag(&act[m * ST_A + 64], klo);
            v16h a3 = ldfrag(&act[m * ST_A + 96], klo);
            const _Float16* wr0 = &w2T[m * ST_W];
            v16h b0 = ldfrag(wr0 +  0, klo), b1f = ldfrag(wr0 + 32, klo);
            v16h b2f = ldfrag(wr0 + 64, klo), b3f = ldfrag(wr0 + 96, klo);
            v8f acc[8];
            #pragma unroll
            for (int t = 0; t < 8; ++t) {
                v16h n0 = b0, n1 = b1f, n2 = b2f, n3 = b3f;
                if (t < 7) {   // prefetch next tile's B while current WMMAs run
                    const _Float16* wn = &w2T[((t + 1) * 16 + m) * ST_W];
                    n0 = ldfrag(wn +  0, klo); n1 = ldfrag(wn + 32, klo);
                    n2 = ldfrag(wn + 64, klo); n3 = ldfrag(wn + 96, klo);
                }
                v8f c = {};
                c = __builtin_amdgcn_wmma_f32_16x16x32_f16(false, a0, false, b0,  (short)0, c, false, false);
                c = __builtin_amdgcn_wmma_f32_16x16x32_f16(false, a1, false, b1f, (short)0, c, false, false);
                c = __builtin_amdgcn_wmma_f32_16x16x32_f16(false, a2, false, b2f, (short)0, c, false, false);
                c = __builtin_amdgcn_wmma_f32_16x16x32_f16(false, a3, false, b3f, (short)0, c, false, false);
                acc[t] = c;
                b0 = n0; b1f = n1; b2f = n2; b3f = n3;
            }
            #pragma unroll
            for (int t = 0; t < 8; ++t) {
                int n = t * 16 + m;
                float bv = b2s[n];
                #pragma unroll
                for (int i = 0; i < 8; ++i) {
                    float v = fmaxf(acc[t][i] + bv, 0.0f);
                    act[(mo + i) * ST_A + n] = (_Float16)v;  // C layout: vgpr i -> row mo+i
                }
            }
        }

        // ---- layer 3 (WMMA) with fused layer-4 epilogue: rs[i] += relu(c+b3)*w4 ----
        float rs[8] = {0.f, 0.f, 0.f, 0.f, 0.f, 0.f, 0.f, 0.f};
        {
            v16h a0 = ldfrag(&act[m * ST_A +  0], klo);
            v16h a1 = ldfrag(&act[m * ST_A + 32], klo);
            v16h a2 = ldfrag(&act[m * ST_A + 64], klo);
            v16h a3 = ldfrag(&act[m * ST_A + 96], klo);
            const _Float16* wr0 = &w3T[m * ST_W];
            v16h b0 = ldfrag(wr0 +  0, klo), b1f = ldfrag(wr0 + 32, klo);
            v16h b2f = ldfrag(wr0 + 64, klo), b3f = ldfrag(wr0 + 96, klo);
            #pragma unroll
            for (int t = 0; t < 8; ++t) {
                v16h n0 = b0, n1 = b1f, n2 = b2f, n3 = b3f;
                if (t < 7) {
                    const _Float16* wn = &w3T[((t + 1) * 16 + m) * ST_W];
                    n0 = ldfrag(wn +  0, klo); n1 = ldfrag(wn + 32, klo);
                    n2 = ldfrag(wn + 64, klo); n3 = ldfrag(wn + 96, klo);
                }
                v8f c = {};
                c = __builtin_amdgcn_wmma_f32_16x16x32_f16(false, a0, false, b0,  (short)0, c, false, false);
                c = __builtin_amdgcn_wmma_f32_16x16x32_f16(false, a1, false, b1f, (short)0, c, false, false);
                c = __builtin_amdgcn_wmma_f32_16x16x32_f16(false, a2, false, b2f, (short)0, c, false, false);
                c = __builtin_amdgcn_wmma_f32_16x16x32_f16(false, a3, false, b3f, (short)0, c, false, false);
                int n = t * 16 + m;
                float bv = b3s[n], wv = w4s[n];
                #pragma unroll
                for (int i = 0; i < 8; ++i)
                    rs[i] = fmaf(fmaxf(c[i] + bv, 0.0f), wv, rs[i]);
                b0 = n0; b1f = n1; b2f = n2; b3f = n3;
            }
        }

        // ---- reduce rows across each 16-lane half (all 32 lanes active) ----
        #pragma unroll
        for (int i = 0; i < 8; ++i) {
            float v = rs[i];
            v += __shfl_xor(v, 1, 32);
            v += __shfl_xor(v, 2, 32);
            v += __shfl_xor(v, 4, 32);
            v += __shfl_xor(v, 8, 32);
            rs[i] = v;
        }

        // ---- ELU + 1, quadrature weight, atomic scatter (lanes 0 and 16) ----
        if (m == 0) {
            const int  rowbase = (lane >> 4) * 8;  // lane 0 -> rows 0-7, lane 16 -> rows 8-15
            const float bb = b4s[0];
            #pragma unroll
            for (int i = 0; i < 8; ++i) {
                long rr = tbase + rowbase + i;
                if (rr < total) {
                    float v  = rs[i] + bb;
                    float dz = (v > 0.0f) ? v : (expf(v) - 1.0f);
                    dz += 1.0f;
                    int   bi = (int)(rr / S_PTS);
                    int   si = (int)(rr - (long)bi * S_PTS);
                    float contrib = dz * ccws[si] * x[bi] * 0.5f;
                    atomicAdd(&out[bi], contrib);   // global_atomic_add_f32
                }
            }
        }
    }
}

extern "C" void kernel_launch(void* const* d_in, const int* in_sizes, int n_in,
                              void* d_out, int out_size, void* d_ws, size_t ws_size,
                              hipStream_t stream) {
    const float* x  = (const float*)d_in[0];
    const float* h  = (const float*)d_in[1];
    const float* w1 = (const float*)d_in[2];
    const float* b1 = (const float*)d_in[3];
    const float* w2 = (const float*)d_in[4];
    const float* b2 = (const float*)d_in[5];
    const float* w3 = (const float*)d_in[6];
    const float* b3 = (const float*)d_in[7];
    const float* w4 = (const float*)d_in[8];
    const float* b4 = (const float*)d_in[9];
    float* out = (float*)d_out;
    const int Bn = in_sizes[0];

    monotonic_init<<<(Bn + 255) / 256, 256, 0, stream>>>(h, out, Bn);

    const long total  = (long)Bn * S_PTS;
    const int  tiles  = (int)((total + 15) / 16);
    // persistent blocks: each wave handles ~TILES_PER_WAVE tiles, amortizing LDS setup
    const int  blocks = (tiles + WAVES * TILES_PER_WAVE - 1) / (WAVES * TILES_PER_WAVE);
    monotonic_main<<<blocks, TPB, 0, stream>>>(x, h, w1, b1, w2, b2, w3, b3, w4, b4, out, Bn);
}
